// LSTMLayer_20461224198665
// MI455X (gfx1250) — matrix operand order repaired and checked by reading, compiled, not run on hardware
//
#include <hip/hip_runtime.h>

#define T_STEPS 512
#define B_DIM   64
#define I_DIM   512
#define H_DIM   512
#define G4      2048   // 4*H
#define KTOT    1024   // I + H

typedef __attribute__((ext_vector_type(16))) _Float16 v16h;
typedef __attribute__((ext_vector_type(8)))  _Float16 v8h;
typedef __attribute__((ext_vector_type(4)))  _Float16 v4h;
typedef __attribute__((ext_vector_type(8)))  float    v8f;

// ---------------------------------------------------------------------------
// Kernel 1: pack [W;U] (f32, row-major [K,4H]) into f16 K-major image
// Gt[n][k]: n in [0,2048) (gate*512 + h), k in [0,1024) (k<512 -> W, else U).
// 4 MB, L2-resident for all 512 steps.
// ---------------------------------------------------------------------------
__global__ __launch_bounds__(256) void pack_weights(const float* __restrict__ W,
                                                    const float* __restrict__ U,
                                                    _Float16* __restrict__ Gt) {
    int idx = blockIdx.x * 256 + threadIdx.x;  // G4 blocks * 256 threads
    int n   = idx & (G4 - 1);
    int k0  = (idx >> 11) * 4;                 // 256 * 4 = 1024 k values
    #pragma unroll
    for (int j = 0; j < 4; ++j) {
        int k   = k0 + j;
        float v = (k < I_DIM) ? W[(size_t)k * G4 + n]
                              : U[(size_t)(k - I_DIM) * G4 + n];
        Gt[(size_t)n * KTOT + k] = (_Float16)v;
    }
}

// ---------------------------------------------------------------------------
// Kernel 2: generic f32 -> f16 conversion, 4 elements / thread.
// ---------------------------------------------------------------------------
__global__ __launch_bounds__(256) void cvt_f32_f16(const float* __restrict__ src,
                                                   _Float16* __restrict__ dst) {
    size_t e = ((size_t)blockIdx.x * 256 + threadIdx.x) * 4;
    const float4 v = *(const float4*)(src + e);
    v4h hv;
    hv[0] = (_Float16)v.x; hv[1] = (_Float16)v.y;
    hv[2] = (_Float16)v.z; hv[3] = (_Float16)v.w;
    *(v4h*)(dst + e) = hv;
}

__device__ __forceinline__ v16h ldb(const _Float16* p) { return *(const v16h*)p; }
__device__ __forceinline__ v8h  lda(const _Float16* p) { return *(const v8h*)p;  }

// ---------------------------------------------------------------------------
// Kernel 3: one recurrence step (no LDS, no barriers).
//   g = [x_t || a_{t-1}] @ Gt^T + b      (64 x 2048, K = 1024, f16 WMMA)
//   a_t = sigmoid(g_o) * tanh(sigmoid(g_u)*tanh(g_c) + sigmoid(g_f)*a_{t-1})
// Grid: 16 WGs x 256 threads (8 waves). Wave w: batch tile mi = w&3,
// h-tile hs = w>>2; owns all 4 gate slices -> gate math is wave-local.
// Single fused 32-tile K loop; distance-3 software pipeline held in place
// with sched_barrier(0) so the scheduler cannot sink prefetched loads back
// to their wmma consumers (which would collapse the pipeline registers).
// ---------------------------------------------------------------------------
__global__ __launch_bounds__(256, 1) void lstm_step(
        const _Float16* __restrict__ x16t,
        const _Float16* __restrict__ af_in,
        _Float16* __restrict__ af_out,
        const float* __restrict__ aprev32,
        const float* __restrict__ bias,
        const _Float16* __restrict__ Gt,
        float* __restrict__ at) {
    const int tid      = threadIdx.x;
    const int wave     = tid >> 5;
    const int lane     = tid & 31;
    const int l16      = lane & 15;
    const int half_sel = (lane < 16) ? 0 : 1;
    const int mi       = wave & 3;
    const int hs       = wave >> 2;
    const int h        = blockIdx.x * 32 + hs * 16 + l16;  // column within H
    const int arow     = mi * 16 + l16;                    // batch row

    // Bias folded into accumulator init (same column for all 8 C rows/lane).
    v8f acc[4];
    #pragma unroll
    for (int g = 0; g < 4; ++g) {
        float bv = bias[g * H_DIM + h];
        #pragma unroll
        for (int i = 0; i < 8; ++i) acc[g][i] = bv;
    }

    // Per-lane fragment base pointers (A 16x32 layout: lanes 0-15 K=k0..15,
    // lanes 16-31 K=k0+16..31; B: 16 contiguous K halves/lane).
    const _Float16* a0p = x16t  + arow * I_DIM + half_sel * 16;   // kk 0..15
    const _Float16* a1p = af_in + arow * H_DIM + half_sel * 16;   // kk 16..31
    const _Float16* bp0 = Gt + (size_t)(0 * H_DIM + h) * KTOT + half_sel * 16;
    const _Float16* bp1 = Gt + (size_t)(1 * H_DIM + h) * KTOT + half_sel * 16;
    const _Float16* bp2 = Gt + (size_t)(2 * H_DIM + h) * KTOT + half_sel * 16;
    const _Float16* bp3 = Gt + (size_t)(3 * H_DIM + h) * KTOT + half_sel * 16;

    #define A_ADDR(kk) ((kk) < 16 ? a0p + (kk) * 32 : a1p + ((kk) - 16) * 32)

    // Pipeline prologue: stages for kk = 0, 1, 2.
    v8h  AL[3], AH[3];
    v16h B0[3], B1[3], B2[3], B3[3];
    #pragma unroll
    for (int i = 0; i < 3; ++i) {
        AL[i] = lda(A_ADDR(i));
        AH[i] = lda(A_ADDR(i) + 8);
        B0[i] = ldb(bp0 + i * 32);
        B1[i] = ldb(bp1 + i * 32);
        B2[i] = ldb(bp2 + i * 32);
        B3[i] = ldb(bp3 + i * 32);
    }

    #pragma unroll
    for (int kk = 0; kk < 32; ++kk) {
        // Issue loads for kk+3 (stay ahead of the wmmas below).
        v8h  aLp{}, aHp{};
        v16h b0p{}, b1p{}, b2p{}, b3p{};
        if (kk + 3 < 32) {
            const int nn = kk + 3;
            aLp = lda(A_ADDR(nn));
            aHp = lda(A_ADDR(nn) + 8);
            b0p = ldb(bp0 + nn * 32);
            b1p = ldb(bp1 + nn * 32);
            b2p = ldb(bp2 + nn * 32);
            b3p = ldb(bp3 + nn * 32);
        }
        __builtin_amdgcn_sched_barrier(0);   // loads(kk+3) stay before wmmas(kk)

        v16h af;
        #pragma unroll
        for (int i = 0; i < 8; ++i) { af[i] = AL[0][i]; af[i + 8] = AH[0][i]; }
        acc[0] = __builtin_amdgcn_wmma_f32_16x16x32_f16(false, af, false, B0[0],
                                                        (short)0, acc[0], false, false);
        acc[1] = __builtin_amdgcn_wmma_f32_16x16x32_f16(false, af, false, B1[0],
                                                        (short)0, acc[1], false, false);
        acc[2] = __builtin_amdgcn_wmma_f32_16x16x32_f16(false, af, false, B2[0],
                                                        (short)0, acc[2], false, false);
        acc[3] = __builtin_amdgcn_wmma_f32_16x16x32_f16(false, af, false, B3[0],
                                                        (short)0, acc[3], false, false);
        __builtin_amdgcn_sched_barrier(0);   // wmmas(kk) stay before loads(kk+4)

        // Rotate pipeline stages (SSA renames).
        AL[0] = AL[1]; AL[1] = AL[2]; AL[2] = aLp;
        AH[0] = AH[1]; AH[1] = AH[2]; AH[2] = aHp;
        B0[0] = B0[1]; B0[1] = B0[2]; B0[2] = b0p;
        B1[0] = B1[1]; B1[1] = B1[2]; B1[2] = b1p;
        B2[0] = B2[1]; B2[1] = B2[2]; B2[2] = b2p;
        B3[0] = B3[1]; B3[1] = B3[2]; B3[2] = b3p;
    }
    #undef A_ADDR

    // Gate combine, wave-local.
    // C/D layout: VGPR r -> M = r (lanes 0-15) or r+8 (lanes 16-31), N = l16.
    const int mbase = mi * 16 + half_sel * 8;
    #pragma unroll
    for (int r = 0; r < 8; ++r) {
        const int   m  = mbase + r;
        const float gu = acc[0][r], gf = acc[1][r], go = acc[2][r], gc = acc[3][r];
        const float ap = aprev32[m * H_DIM + h];
        const float su = 1.0f / (1.0f + __expf(-gu));
        const float sf = 1.0f / (1.0f + __expf(-gf));
        const float so = 1.0f / (1.0f + __expf(-go));
        const float ct = 2.0f / (1.0f + __expf(-2.0f * gc)) - 1.0f;  // tanh(gc)
        const float c  = su * ct + sf * ap;   // forget gate * prev HIDDEN state
        const float th = 2.0f / (1.0f + __expf(-2.0f * c)) - 1.0f;   // tanh(c)
        const float a  = so * th;
        at[m * H_DIM + h]     = a;             // f32 output (and exact state)
        af_out[m * H_DIM + h] = (_Float16)a;   // f16 state for next step's A
    }
}

// ---------------------------------------------------------------------------
extern "C" void kernel_launch(void* const* d_in, const int* in_sizes, int n_in,
                              void* d_out, int out_size, void* d_ws, size_t ws_size,
                              hipStream_t stream) {
    (void)in_sizes; (void)n_in; (void)out_size; (void)ws_size;
    const float* x  = (const float*)d_in[0];  // [T,B,I]
    const float* a0 = (const float*)d_in[1];  // [B,H]
    const float* W  = (const float*)d_in[2];  // [I,4H]
    const float* U  = (const float*)d_in[3];  // [H,4H]
    const float* b  = (const float*)d_in[4];  // [4H]
    float* out = (float*)d_out;               // [T,B,H]

    // Workspace layout (36.2 MB):
    //   Gt : f16 [2048][1024]           4 MB
    //   x16: f16 [T][B][I]             32 MB
    //   af : f16 [2][B][H]            128 KB  (state ping-pong)
    char* ws = (char*)d_ws;
    _Float16* Gt  = (_Float16*)ws;
    _Float16* x16 = (_Float16*)(ws + (size_t)G4 * KTOT * 2);
    _Float16* af  = (_Float16*)(ws + (size_t)G4 * KTOT * 2
                                   + (size_t)T_STEPS * B_DIM * I_DIM * 2);

    pack_weights<<<G4, 256, 0, stream>>>(W, U, Gt);
    // x: 512*64*512 = 16,777,216 elems -> 16384 blocks * 256 thr * 4
    cvt_f32_f16<<<16384, 256, 0, stream>>>(x, x16);
    // a0: 32768 elems -> 32 blocks
    cvt_f32_f16<<<32, 256, 0, stream>>>(a0, af);

    for (int t = 0; t < T_STEPS; ++t) {
        const _Float16* xt  = x16 + (size_t)t * B_DIM * I_DIM;
        const _Float16* ain = af + (size_t)(t & 1) * B_DIM * H_DIM;
        _Float16*       aot = af + (size_t)((t + 1) & 1) * B_DIM * H_DIM;
        const float*    ap  = (t == 0) ? a0 : out + (size_t)(t - 1) * B_DIM * H_DIM;
        lstm_step<<<16, 256, 0, stream>>>(xt, ain, aot, ap, b, Gt,
                                          out + (size_t)t * B_DIM * H_DIM);
    }
}
